// RNN_50242527429326
// MI455X (gfx1250) — matrix operand order repaired and checked by reading, not compiled, not run on hardware
//
#include <hip/hip_runtime.h>

#define INPUT    28
#define HIDDEN   64
#define GATES    256   // 4*HIDDEN
#define OUTD     10
#define BTILE    64
#define NTHREADS 256

typedef __attribute__((ext_vector_type(16))) _Float16 v16h;
typedef __attribute__((ext_vector_type(8)))  _Float16 v8h;
typedef __attribute__((ext_vector_type(8)))  float    v8f;
typedef __attribute__((ext_vector_type(4)))  float    v4f;

// 16-half WMMA fragment slice: two 16-byte LDS loads (ds_load_b128 each).
__device__ __forceinline__ v16h ld_frag32(const _Float16* p) {
  v8h lo = *(const v8h*)(p);
  v8h hi = *(const v8h*)(p + 16);
  v16h r;
#pragma unroll
  for (int i = 0; i < 8; ++i) { r[i] = lo[i]; r[i + 8] = hi[i]; }
  return r;
}

// gfx1250: single-op v_tanh_f32; fallback for other targets / host pass.
#if defined(__AMDGCN__) && __has_builtin(__builtin_amdgcn_tanhf)
__device__ __forceinline__ float fast_tanh(float x) { return __builtin_amdgcn_tanhf(x); }
#else
__device__ __forceinline__ float fast_tanh(float x) {
  return 1.0f - 2.0f * __builtin_amdgcn_rcpf(__expf(2.0f * x) + 1.0f);
}
#endif
// Gate rows for i/f/o are pre-scaled by 0.5, so acc == z/2 already:
__device__ __forceinline__ float sigmoid_pre(float half_z) {
  return __builtin_fmaf(0.5f, fast_tanh(half_z), 0.5f);
}

// Per-lane x A-fragment gather: 16 floats (K = kh*8..+7 and 16+kh*8..+7),
// cols >= 28 zeroed. Row base is 112B = 16B aligned -> aligned b128 loads.
// Branchless: q[5] (cols 20..23) is always in-bounds; select zero for kh=1.
__device__ __forceinline__ void load_xfrag(const float* xrowp, int kh,
                                           v4f& a, v4f& b, v4f& c, v4f& d) {
  const v4f* q = (const v4f*)xrowp;
  a = q[kh * 2 + 0];          // cols kh*8   .. kh*8+3
  b = q[kh * 2 + 1];          // cols kh*8+4 .. kh*8+7
  c = q[kh * 2 + 4];          // kh=0: cols 16..19 ; kh=1: cols 24..27
  v4f d5 = q[5];              // cols 20..23 (always valid)
  d = kh ? (v4f){0.0f, 0.0f, 0.0f, 0.0f} : d5;   // kh=1 -> pad cols 28..31
}

__global__ __launch_bounds__(NTHREADS, 1)
void lstm_fused(const float* __restrict__ x,
                const float* __restrict__ Wih,
                const float* __restrict__ Whh,
                const float* __restrict__ bih,
                const float* __restrict__ bhh,
                const float* __restrict__ Wfc,
                const float* __restrict__ bfc,
                float* __restrict__ out,
                int T) {
  // 48KB region: weight staging during init, then double-buffered h state.
  __shared__ __align__(32) char smem[49152];
  __shared__ float sBias[GATES];

  const int tid  = threadIdx.x;
  const int lane = tid & 31;
  const int wave = tid >> 5;
  const int nlo  = lane & 15;    // N (or M for A-frags) within tile
  const int kh   = lane >> 4;    // K half select
  const int m    = wave >> 1;    // M-tile 0..3 (16 batch rows each)
  const int p    = wave & 1;     // hidden half (cols 0..31 vs 32..63)
  const int bbase = blockIdx.x * BTILE;

  // ---- phase 1: stage weights into LDS as f16 (i/f/o rows pre-scaled 0.5) ----
  _Float16* sWih = (_Float16*)smem;              // [256][32] = 16KB
  _Float16* sWhh = (_Float16*)(smem + 16384);    // [256][64] = 32KB
  for (int idx = tid; idx < GATES * 32; idx += NTHREADS) {
    int r = idx >> 5, c = idx & 31;
    float scale = ((r >> 6) == 2) ? 1.0f : 0.5f;   // g-gate rows unscaled
    sWih[idx] = (_Float16)(c < INPUT ? Wih[r * INPUT + c] * scale : 0.0f);
  }
  for (int idx = tid; idx < GATES * HIDDEN; idx += NTHREADS) {
    int r = idx >> 6;
    float scale = ((r >> 6) == 2) ? 1.0f : 0.5f;
    sWhh[idx] = (_Float16)(Whh[idx] * scale);
  }
  if (tid < GATES) {
    float scale = ((tid >> 6) == 2) ? 1.0f : 0.5f;
    sBias[tid] = (bih[tid] + bhh[tid]) * scale;
  }
  __syncthreads();

  // ---- phase 2: pin loop-invariant B fragments + bias in VGPRs ----
  v16h bWih[8], bWhh0[8], bWhh1[8];
  float biasv[8];
#pragma unroll
  for (int g4 = 0; g4 < 4; ++g4) {
#pragma unroll
    for (int hcc = 0; hcc < 2; ++hcc) {
      int tt = g4 * 2 + hcc;
      int nt = g4 * 4 + (2 * p + hcc);
      bWih[tt]  = ld_frag32(&sWih[(nt * 16 + nlo) * 32 + kh * 8]);
      bWhh0[tt] = ld_frag32(&sWhh[(nt * 16 + nlo) * HIDDEN + kh * 8]);
      bWhh1[tt] = ld_frag32(&sWhh[(nt * 16 + nlo) * HIDDEN + 32 + kh * 8]);
      biasv[tt] = sBias[nt * 16 + nlo];
    }
  }
  __syncthreads();   // weight region dead; reuse as h buffers

  // ---- phase 3: steady state — double-buffered h in reused LDS ----
  _Float16* hA = (_Float16*)smem;                // 8KB
  _Float16* hB = (_Float16*)(smem + 8192);       // 8KB
  for (int idx = tid; idx < BTILE * HIDDEN; idx += NTHREADS)
    hA[idx] = (_Float16)0.0f;                    // h_0 = 0

  const float* xcur = x + (size_t)(bbase + m * 16 + nlo) * T * INPUT;  // x_t row
  v4f xa, xb, xc, xd;
  load_xfrag(xcur, kh, xa, xb, xc, xd);          // t = 0

  v8f cst[2];
#pragma unroll
  for (int h = 0; h < 2; ++h)
#pragma unroll
    for (int r = 0; r < 8; ++r) cst[h][r] = 0.0f;

  const int hoff0 = (m * 16 + nlo) * HIDDEN + kh * 8;   // A-frag base offset

  auto step = [&](int t, const _Float16* hcur, _Float16* hnxt) {
    __syncthreads();   // h_t visible; prior reads of hnxt's buffer complete

    // issue global loads of x_{t+1} fragment under this step's compute
    v4f na, nb, nc, nd;
    if (t + 1 < T) {
      load_xfrag(xcur + INPUT, kh, na, nb, nc, nd);
      __builtin_prefetch(xcur + 2 * INPUT, 0, 1);   // speculative t+2
    } else {
      na = xa; nb = xb; nc = xc; nd = xd;
    }

    // x_t fragment -> f16
    v16h ax;
#pragma unroll
    for (int i = 0; i < 4; ++i) {
      ax[i]      = (_Float16)xa[i];
      ax[4 + i]  = (_Float16)xb[i];
      ax[8 + i]  = (_Float16)xc[i];
      ax[12 + i] = (_Float16)xd[i];
    }

    // h_t A-fragments from LDS
    v16h ah0 = ld_frag32(&hcur[hoff0]);
    v16h ah1 = ld_frag32(&hcur[hoff0 + 32]);

    // z(/2 for i,f,o) = bias + x_t @ Wih^T + h_t @ Whh^T : 8 tiles x 3 WMMAs
    v8f acc[8];
#pragma unroll
    for (int tt = 0; tt < 8; ++tt) {
      v8f a;
#pragma unroll
      for (int r = 0; r < 8; ++r) a[r] = biasv[tt];
      a = __builtin_amdgcn_wmma_f32_16x16x32_f16(false, ax,  false, bWih[tt],  (short)0, a, false, false);
      a = __builtin_amdgcn_wmma_f32_16x16x32_f16(false, ah0, false, bWhh0[tt], (short)0, a, false, false);
      a = __builtin_amdgcn_wmma_f32_16x16x32_f16(false, ah1, false, bWhh1[tt], (short)0, a, false, false);
      acc[tt] = a;
    }

    // gates (i,f,g,o) + cell update, publish h_{t+1}
#pragma unroll
    for (int hcc = 0; hcc < 2; ++hcc) {
      int col = (2 * p + hcc) * 16 + nlo;
#pragma unroll
      for (int r = 0; r < 8; ++r) {
        float iv = sigmoid_pre(acc[0 + hcc][r]);
        float fv = sigmoid_pre(acc[2 + hcc][r]);
        float gv = fast_tanh  (acc[4 + hcc][r]);
        float ov = sigmoid_pre(acc[6 + hcc][r]);
        float c  = fv * cst[hcc][r] + iv * gv;
        cst[hcc][r] = c;
        float h  = ov * fast_tanh(c);
        hnxt[(m * 16 + r + 8 * kh) * HIDDEN + col] = (_Float16)h;
      }
    }

    // rotate x pipeline
    xa = na; xb = nb; xc = nc; xd = nd;
    xcur += INPUT;
  };

  // T-loop unrolled by 2: static buffer roles, no pointer swap
  for (int t = 0; t + 1 < T; t += 2) {
    step(t,     hA, hB);
    step(t + 1, hB, hA);
  }
  const _Float16* hfin = hA;
  if (T & 1) { step(T - 1, hA, hB); hfin = hB; }
  __syncthreads();   // h_T complete

  // ---- final FC: out = h_T @ Wfc^T + bfc ----
  for (int idx = tid; idx < BTILE * OUTD; idx += NTHREADS) {
    int row = idx / OUTD, col = idx % OUTD;
    float s = bfc[col];
#pragma unroll 8
    for (int k = 0; k < HIDDEN; ++k)
      s += (float)hfin[row * HIDDEN + k] * Wfc[col * HIDDEN + k];
    out[(size_t)(bbase + row) * OUTD + col] = s;
  }
}

extern "C" void kernel_launch(void* const* d_in, const int* in_sizes, int n_in,
                              void* d_out, int out_size, void* d_ws, size_t ws_size,
                              hipStream_t stream) {
  (void)n_in; (void)d_ws; (void)ws_size;
  const float* x   = (const float*)d_in[0];
  const float* Wih = (const float*)d_in[1];
  const float* Whh = (const float*)d_in[2];
  const float* bih = (const float*)d_in[3];
  const float* bhh = (const float*)d_in[4];
  const float* Wfc = (const float*)d_in[5];
  const float* bfc = (const float*)d_in[6];
  float* out = (float*)d_out;

  int B = out_size / OUTD;                 // 8192
  int T = in_sizes[0] / (B * INPUT);       // 128

  dim3 grid(B / BTILE), block(NTHREADS);
  hipLaunchKernelGGL(lstm_fused, grid, block, 0, stream,
                     x, Wih, Whh, bih, bhh, Wfc, bfc, out, T);
}
